// BahdanauAttentionLayer_60988535604017
// MI455X (gfx1250) — matrix operand order repaired and checked by reading, not compiled, not run on hardware
//
#include <hip/hip_runtime.h>
#include <hip/hip_bf16.h>
#include <stdint.h>

// Problem constants (from reference)
#define SRCLEN  1024
#define BSZ     64
#define IN_DIM  1024
#define OUT_DIM 1024
#define KDIM2   1024                 // K of the hot GEMM after hoisting W1*input
#define MTOT    (SRCLEN * BSZ)       // 65536

// WMMA vector types (probe-confirmed signatures)
typedef __attribute__((ext_vector_type(16))) __bf16 v16bf;
typedef __attribute__((ext_vector_type(8)))  float  v8f;

union FragB16 { v16bf v; uint4 q[2]; };

__device__ __forceinline__ unsigned short f2bf(float x) {
    unsigned int u = __float_as_uint(x);
    unsigned int r = u + 0x7FFFu + ((u >> 16) & 1u);  // round-to-nearest-even
    return (unsigned short)(r >> 16);
}

__device__ __forceinline__ float fast_tanh(float x) {
#if __has_builtin(__builtin_amdgcn_tanhf)
    return __builtin_amdgcn_tanhf(x);       // v_tanh_f32 (gfx1250 TRANS op)
#elif __has_builtin(__builtin_amdgcn_tanh_f32)
    return __builtin_amdgcn_tanh_f32(x);
#else
    return tanhf(x);
#endif
}

// ---------------------------------------------------------------------------
// K0a: P[b][o] = sum_{k<1024} input[b][k] * W_in[o][k]   (f32, s-invariant)
// Stored transposed: Pt[o][b] so the WMMA kernel can C-init with b128 loads.
// Grid 256, block 256: thread -> (o = blk*4 + t>>6, b = t&63), K via float4.
// ---------------------------------------------------------------------------
__global__ __launch_bounds__(256)
void k_precompute_p(const float* __restrict__ input,   // [64][1024]
                    const float* __restrict__ Win,     // [1024][2048]
                    float* __restrict__ Pt) {          // [1024][64]
    const int t = threadIdx.x;
    const int o = blockIdx.x * 4 + (t >> 6);
    const int b = t & 63;
    const float* irow = input + (size_t)b * IN_DIM;
    const float* wrow = Win + (size_t)o * (IN_DIM + OUT_DIM);   // first 1024 = W1
    float acc = 0.0f;
#pragma unroll 4
    for (int k = 0; k < IN_DIM; k += 4) {
        float4 a = *(const float4*)(irow + k);
        float4 w = *(const float4*)(wrow + k);
        acc = fmaf(a.x, w.x, acc);
        acc = fmaf(a.y, w.y, acc);
        acc = fmaf(a.z, w.z, acc);
        acc = fmaf(a.w, w.w, acc);
    }
    Pt[(size_t)o * BSZ + b] = acc;
}

// ---------------------------------------------------------------------------
// K0b: W2 = W_in[:, 1024:2048] (f32) -> bf16 [1024][1024] in workspace
// ---------------------------------------------------------------------------
__global__ void k_convert_w2(const float* __restrict__ Win,
                             unsigned short* __restrict__ Wb2) {
    int idx = blockIdx.x * blockDim.x + threadIdx.x;      // 262144 float4s
    int o   = idx >> 8;              // /256 float4s per row
    int c   = (idx & 255) << 2;
    float4 v = *(const float4*)(Win + (size_t)o * (IN_DIM + OUT_DIM) + IN_DIM + c);
    ushort4 h;
    h.x = f2bf(v.x); h.y = f2bf(v.y); h.z = f2bf(v.z); h.w = f2bf(v.w);
    *(ushort4*)(Wb2 + (size_t)o * KDIM2 + c) = h;
}

// ---------------------------------------------------------------------------
// K1: fused GEMM (bf16 WMMA, f32 acc, C-init = Pt) + tanh + W_v dot + mask
//
// One workgroup = one s (all 64 b rows), 8 waves. K = 1024 (src part only).
//   * A-tile: 64 x 1024 bf16 (+pad) in LDS = 129 KB -> 2 WGs/WGP.
//   * Each wave owns 128 N columns; 4 accumulators (4 M-subtiles of 16 rows)
//     reuse every B fragment pulled from L2 across 4 WMMAs.
// ---------------------------------------------------------------------------
#define MT     64                      // rows per workgroup (= BSZ, one s)
#define APAD   8                       // +16B per row: bank-conflict-free b128
#define AROW   (KDIM2 + APAD)          // 1032 halfs per LDS row

__global__ __launch_bounds__(256)
void k_score(const float* __restrict__ src,          // [1024][64][1024]
             const unsigned char* __restrict__ mask, // [1024][64] (bool bytes)
             const unsigned short* __restrict__ Wb2, // bf16 [1024][1024]
             const float* __restrict__ Pt,           // f32 [1024][64]
             const float* __restrict__ Wv,           // [1024]
             float* __restrict__ logits) {           // [65536]
    __shared__ __align__(16) unsigned short Als[MT * AROW];   // 129 KB
    __shared__ float sc[MT];

    const int tid = threadIdx.x;
    const int s   = blockIdx.x;

    // ---- build bf16 A-tile (64 x 1024 = source_hids rows for this s)
    for (int i = tid; i < MT * (KDIM2 / 4); i += 256) {   // 16384 float4s
        int row = i >> 8;            // /256
        int col = (i & 255) << 2;
        float4 v = *(const float4*)(src + ((size_t)s * BSZ + row) * OUT_DIM + col);
        ushort4 h;
        h.x = f2bf(v.x); h.y = f2bf(v.y); h.z = f2bf(v.z); h.w = f2bf(v.w);
        *(ushort4*)&Als[row * AROW + col] = h;
    }
    if (tid < MT) sc[tid] = 0.0f;
    __syncthreads();

    const int wave = tid >> 5;
    const int lane = tid & 31;
    const int ln   = lane & 15;
    const int half = lane >> 4;

    float part[4][8];
#pragma unroll
    for (int sub = 0; sub < 4; ++sub)
#pragma unroll
        for (int r = 0; r < 8; ++r) part[sub][r] = 0.0f;

    for (int nt = 0; nt < 8; ++nt) {
        const int ocol = wave * 128 + nt * 16 + ln;       // this lane's N column
        const unsigned short* wrow = Wb2 + (size_t)ocol * KDIM2;
        const float*          prow = Pt  + (size_t)ocol * BSZ;

        // C-init from precomputed f32 input-half: c[sub][r] = P[b][ocol],
        // b = sub*16 + half*8 + r  -> 8 contiguous floats of Pt row.
        v8f c[4];
#pragma unroll
        for (int sub = 0; sub < 4; ++sub) {
            float4 p0 = *(const float4*)(prow + sub * 16 + half * 8);
            float4 p1 = *(const float4*)(prow + sub * 16 + half * 8 + 4);
            c[sub][0] = p0.x; c[sub][1] = p0.y; c[sub][2] = p0.z; c[sub][3] = p0.w;
            c[sub][4] = p1.x; c[sub][5] = p1.y; c[sub][6] = p1.z; c[sub][7] = p1.w;
        }

#pragma unroll 2
        for (int k0 = 0; k0 < KDIM2; k0 += 32) {
            // B frag (32x16: lane half selects contiguous 16-K slab) -- L2 hit
            FragB16 b;
            const unsigned short* bp = wrow + k0 + half * 16;
            b.q[0] = *(const uint4*)(bp);
            b.q[1] = *(const uint4*)(bp + 8);
            // 4 M-subtiles reuse this B fragment
#pragma unroll
            for (int sub = 0; sub < 4; ++sub) {
                FragB16 a;
                const unsigned short* ap =
                    &Als[(sub * 16 + ln) * AROW + k0 + half * 16];
                a.q[0] = *(const uint4*)(ap);
                a.q[1] = *(const uint4*)(ap + 8);
                c[sub] = __builtin_amdgcn_wmma_f32_16x16x32_bf16(
                             false, a.v, false, b.v, (short)0, c[sub],
                             false, false);
            }
        }
        // fuse tanh + v-dot; C layout: lane half h holds rows M = 8h + r
        const float wv = Wv[ocol];
#pragma unroll
        for (int sub = 0; sub < 4; ++sub)
#pragma unroll
            for (int r = 0; r < 8; ++r)
                part[sub][r] = fmaf(fast_tanh(c[sub][r]), wv, part[sub][r]);
    }

    // reduce across the 16 lanes of each half (xor<16 stays within half)
#pragma unroll
    for (int off = 1; off < 16; off <<= 1)
#pragma unroll
        for (int sub = 0; sub < 4; ++sub)
#pragma unroll
            for (int r = 0; r < 8; ++r)
                part[sub][r] += __shfl_xor(part[sub][r], off, 32);

    if (ln == 0) {
#pragma unroll
        for (int sub = 0; sub < 4; ++sub)
#pragma unroll
            for (int r = 0; r < 8; ++r)
                atomicAdd(&sc[sub * 16 + half * 8 + r], part[sub][r]); // ds_add_f32
    }
    __syncthreads();

    if (tid < MT) {
        float v = sc[tid];
        if (mask[(size_t)s * BSZ + tid]) v = -__builtin_inff();
        logits[(size_t)s * BSZ + tid] = v;
    }
}

// ---------------------------------------------------------------------------
// K2: softmax over s (1024) for each b; also zero-init out[0..65535]
// ---------------------------------------------------------------------------
__global__ __launch_bounds__(256)
void k_softmax(const float* __restrict__ logits,
               float* __restrict__ probs,      // d_out + 65536 (attn_scores)
               float* __restrict__ out0) {     // d_out, zero-init here
    __shared__ float rbuf[8];
    const int b    = blockIdx.x;
    const int tid  = threadIdx.x;
    const int wave = tid >> 5, lane = tid & 31;

    float x[4], mx = -__builtin_inff();
#pragma unroll
    for (int i = 0; i < 4; ++i) {
        x[i] = logits[(size_t)(tid + i * 256) * BSZ + b];
        mx = fmaxf(mx, x[i]);
    }
#pragma unroll
    for (int off = 16; off >= 1; off >>= 1)
        mx = fmaxf(mx, __shfl_xor(mx, off, 32));
    if (lane == 0) rbuf[wave] = mx;
    __syncthreads();
    mx = rbuf[0];
#pragma unroll
    for (int w = 1; w < 8; ++w) mx = fmaxf(mx, rbuf[w]);
    __syncthreads();

    float sum = 0.0f;
#pragma unroll
    for (int i = 0; i < 4; ++i) { x[i] = __expf(x[i] - mx); sum += x[i]; }
#pragma unroll
    for (int off = 16; off >= 1; off >>= 1)
        sum += __shfl_xor(sum, off, 32);
    if (lane == 0) rbuf[wave] = sum;
    __syncthreads();
    float tot = 0.0f;
#pragma unroll
    for (int w = 0; w < 8; ++w) tot += rbuf[w];
    const float inv = __frcp_rn(tot);

#pragma unroll
    for (int i = 0; i < 4; ++i)
        probs[(size_t)(tid + i * 256) * BSZ + b] = x[i] * inv;

    // zero out[b][*] for K3's atomics (64 blocks x 256 thr x 4 = 65536)
    float4 z = {0.f, 0.f, 0.f, 0.f};
    *(float4*)(out0 + (size_t)b * OUT_DIM + tid * 4) = z;
}

// ---------------------------------------------------------------------------
// K3: out[b][o] += sum_s p[s,b] * src[s,b,o]   (grid: 64 b x 8 s-slices)
// ---------------------------------------------------------------------------
__global__ __launch_bounds__(256)
void k_wsum(const float* __restrict__ src,
            const float* __restrict__ probs,
            float* __restrict__ out0) {
    __shared__ float pl[128];
    const int b     = blockIdx.x;
    const int sbase = blockIdx.y * 128;
    const int tid   = threadIdx.x;

    if (tid < 128) pl[tid] = probs[(size_t)(sbase + tid) * BSZ + b];
    __syncthreads();

    const float* p = src + ((size_t)sbase * BSZ + b) * OUT_DIM + tid * 4;
    const size_t stride = (size_t)BSZ * OUT_DIM;
    float4 acc = {0.f, 0.f, 0.f, 0.f};
#pragma unroll 4
    for (int si = 0; si < 128; ++si) {
        __builtin_prefetch(p + 2 * stride, 0, 1);   // global_prefetch_b8
        float4 v = *(const float4*)p;
        const float w = pl[si];
        acc.x = fmaf(w, v.x, acc.x);
        acc.y = fmaf(w, v.y, acc.y);
        acc.z = fmaf(w, v.z, acc.z);
        acc.w = fmaf(w, v.w, acc.w);
        p += stride;
    }
    float* o = out0 + (size_t)b * OUT_DIM + tid * 4;
    atomicAdd(o + 0, acc.x);
    atomicAdd(o + 1, acc.y);
    atomicAdd(o + 2, acc.z);
    atomicAdd(o + 3, acc.w);
}

// ---------------------------------------------------------------------------
extern "C" void kernel_launch(void* const* d_in, const int* in_sizes, int n_in,
                              void* d_out, int out_size, void* d_ws, size_t ws_size,
                              hipStream_t stream) {
    const float*         input = (const float*)d_in[0];          // [64][1024]
    const float*         src   = (const float*)d_in[1];          // [1024][64][1024]
    const unsigned char* mask  = (const unsigned char*)d_in[2];  // bool bytes
    const float*         Win   = (const float*)d_in[3];          // [1024][2048]
    const float*         Wv    = (const float*)d_in[4];          // [1024]

    float* out0  = (float*)d_out;                   // [64][1024]
    float* attn  = out0 + (size_t)BSZ * OUT_DIM;    // [1024][64]

    unsigned short* Wb2    = (unsigned short*)d_ws;                           // 2 MB
    float*          Pt     = (float*)((char*)d_ws + (size_t)2 * 1024 * 1024); // 256 KB
    float*          logits = Pt + (size_t)OUT_DIM * BSZ;                      // 256 KB

    k_precompute_p<<<256, 256, 0, stream>>>(input, Win, Pt);
    k_convert_w2<<<1024, 256, 0, stream>>>(Win, Wb2);
    k_score<<<SRCLEN, 256, 0, stream>>>(src, mask, Wb2, Pt, Wv, logits);
    k_softmax<<<BSZ, 256, 0, stream>>>(logits, attn, out0);
    k_wsum<<<dim3(BSZ, 8), 256, 0, stream>>>(src, attn, out0);
}
